// LSTMConvATTN_63290638074627
// MI455X (gfx1250) — compile-verified, matrix operand order checked by reading, not run on hardware
//
#include <hip/hip_runtime.h>
#include <hip/hip_bf16.h>

// ---------------------------------------------------------------------------
// Types for CDNA5 WMMA (wave32, 16x16x32 bf16 -> f32)
// ---------------------------------------------------------------------------
typedef __attribute__((ext_vector_type(16))) __bf16 v16bf;
typedef __attribute__((ext_vector_type(8)))  __bf16 bf16x8;
typedef __attribute__((ext_vector_type(8)))  float  v8f;

#define B_TOT 64
#define S_LEN 1024
#define HIDD  512
#define GATES 2048
#define KC    576      // 64 (x) + 512 (h)
#define KVLEN 320      // tgt_len (256) + OUT_LEN (64)
#define NHEAD 8
#define DH    64

static __device__ __forceinline__ v8f wmma_bf16(v16bf a, v16bf b, v8f c) {
  return __builtin_amdgcn_wmma_f32_16x16x32_bf16(
      /*neg_a=*/false, a, /*neg_b=*/false, b,
      /*c_mod=*/(short)0, c, /*reuse_a=*/false, /*reuse_b=*/false);
}

// Build a 16-element bf16 fragment from two 8-element (16B) contiguous chunks.
// Matches the ISA's 16-bit A/B layout: lane = (idx%16, half=lane/16),
// chunk1 = K [k0 + half*8 .. +7], chunk2 = K [k0+16+half*8 .. +7].
static __device__ __forceinline__ v16bf ldfrag(const __bf16* p1, const __bf16* p2) {
  bf16x8 lo = *(const bf16x8*)p1;
  bf16x8 hi = *(const bf16x8*)p2;
  return __builtin_shufflevector(lo, hi, 0,1,2,3,4,5,6,7,8,9,10,11,12,13,14,15);
}

static __device__ __forceinline__ float sigf(float x) {
  return 1.0f / (1.0f + __expf(-x));
}
static __device__ __forceinline__ float tanhf_(float x) {
  float e = __expf(2.0f * x);
  return (e - 1.0f) / (e + 1.0f);
}

// ---------------------------------------------------------------------------
// Weight prep: f32 -> bf16 repacks (N-major rows of contiguous K)
// ---------------------------------------------------------------------------
__global__ void prep_wc(const float* __restrict__ Wih, const float* __restrict__ Whh,
                        const float* __restrict__ bih, const float* __restrict__ bhh,
                        __bf16* __restrict__ Wc, float* __restrict__ biasc) {
  int idx = blockIdx.x * 256 + threadIdx.x;
  if (idx < GATES * KC) {
    int n = idx / KC, k = idx % KC;
    float v = (k < 64) ? Wih[n * 64 + k] : Whh[n * HIDD + (k - 64)];
    Wc[idx] = (__bf16)v;
  }
  if (idx < GATES) biasc[idx] = bih[idx] + bhh[idx];
}

__global__ void prep_conv(const float* __restrict__ cw, __bf16* __restrict__ cwB) {
  int idx = blockIdx.x * 256 + threadIdx.x;
  if (idx < HIDD * 3 * HIDD) {
    int o = idx / (3 * HIDD), k = idx % (3 * HIDD);
    int tau = k / HIDD, i = k % HIDD;
    cwB[idx] = (__bf16)cw[(o * HIDD + i) * 3 + tau];
  }
}

// WT[o][i] = W[i*512 + o], o < 512, i < K
__global__ void prep_T(const float* __restrict__ W, __bf16* __restrict__ WT, int K) {
  int idx = blockIdx.x * 256 + threadIdx.x;
  if (idx < HIDD * K) {
    int o = idx / K, i = idx % K;
    WT[idx] = (__bf16)W[(size_t)i * HIDD + o];
  }
}

// WvT_eff[o][i] = Wv[i][o] + Wv[512+i][o]   (because v = concat(kmat, kmat))
__global__ void prep_vT(const float* __restrict__ Wv, __bf16* __restrict__ WvT) {
  int idx = blockIdx.x * 256 + threadIdx.x;
  if (idx < HIDD * HIDD) {
    int o = idx / HIDD, i = idx % HIDD;
    WvT[idx] = (__bf16)(Wv[(size_t)i * HIDD + o] + Wv[(size_t)(HIDD + i) * HIDD + o]);
  }
}

// ---------------------------------------------------------------------------
// Persistent LSTM recurrence. Grid = 4 blocks (16 batches each), 512 threads
// (16 waves). Per step: gates(16x2048) = [x_t | h](16x576) @ Wc^T via WMMA,
// cell state lives in accumulator-layout VGPRs, h goes back to LDS as bf16.
// Weights (2.36 MB) are re-streamed from L2 every step. A laundered *offset*
// (not pointer) keeps LICM from hoisting the 144 B-fragments out of the time
// loop while preserving addrspace inference -> global_load_b128 (LOADcnt
// only), not flat_load (LOADcnt+DScnt).
// ---------------------------------------------------------------------------
__global__ __launch_bounds__(512)
void lstm_kernel(const float* __restrict__ src, const __bf16* __restrict__ Wc,
                 const float* __restrict__ biasc,
                 __bf16* __restrict__ enc, float* __restrict__ hnf,
                 __bf16* __restrict__ hnb) {
  __shared__ __bf16 As[16 * KC];   // rows = batch m, cols = [x(64) | h(512)]

  const int tid  = threadIdx.x;
  const int w    = tid >> 5;       // wave 0..15, owns h-cols [w*32, w*32+32)
  const int ln   = tid & 31;
  const int ln16 = ln & 15;
  const int hh   = ln >> 4;        // half select for fragment layout
  const int b0   = blockIdx.x * 16;

  // zero the h region of A
  for (int i = tid; i < 16 * HIDD; i += 512)
    As[(i >> 9) * KC + 64 + (i & 511)] = (__bf16)0.0f;

  // load x_0
  {
    const int m = tid >> 5, kk = (tid & 31) << 1;
    const float* sp = src + ((size_t)(b0 + m) * S_LEN) * 64 + kk;
    As[m * KC + kk]     = (__bf16)sp[0];
    As[m * KC + kk + 1] = (__bf16)sp[1];
  }

  // per-lane gate biases (i,f,g,o) x (2 n-tiles)
  float bia[4][2];
#pragma unroll
  for (int g = 0; g < 4; ++g)
#pragma unroll
    for (int j = 0; j < 2; ++j)
      bia[g][j] = biasc[g * HIDD + w * 32 + j * 16 + ln16];

  v8f cst[2] = {};                 // cell state, accumulator layout
  __syncthreads();

  const __bf16* arow = As + ln16 * KC;

  for (int t = 0; t < S_LEN; ++t) {
    // Launder a zero offset every iteration: the optimizer cannot prove the
    // B-fragment addresses are t-invariant (so no hoist/spill), but pointer
    // provenance is preserved so the loads stay global_load_b128.
    unsigned long long toff = 0;
    asm volatile("" : "+s"(toff));
    const __bf16* Wct = Wc + toff;

    v8f acc[4][2] = {};

    for (int kt = 0; kt < KC / 32; ++kt) {
      const int c1 = kt * 32 + hh * 8;
      const int c2 = c1 + 16;
      v16bf a = ldfrag(arow + c1, arow + c2);       // from LDS
#pragma unroll
      for (int g = 0; g < 4; ++g) {
#pragma unroll
        for (int j = 0; j < 2; ++j) {
          const __bf16* brow =
              Wct + (size_t)(g * HIDD + w * 32 + j * 16 + ln16) * KC;
          acc[g][j] = wmma_bf16(a, ldfrag(brow + c1, brow + c2), acc[g][j]);
        }
      }
    }
    __syncthreads();   // everyone done reading A for step t

    // epilogue: gate nonlinearities + cell update, all in-register
#pragma unroll
    for (int j = 0; j < 2; ++j) {
      const int hcol = w * 32 + j * 16 + ln16;
#pragma unroll
      for (int r = 0; r < 8; ++r) {
        float iv = sigf(acc[0][j][r] + bia[0][j]);
        float fv = sigf(acc[1][j][r] + bia[1][j]);
        float gv = tanhf_(acc[2][j][r] + bia[2][j]);
        float ov = sigf(acc[3][j][r] + bia[3][j]);
        float c  = fv * cst[j][r] + iv * gv;
        cst[j][r] = c;
        float hv = ov * tanhf_(c);
        const int m = r + (hh << 3);                 // C-fragment row
        As[m * KC + 64 + hcol] = (__bf16)hv;         // h for next step
        if (t < KVLEN)
          enc[((size_t)(b0 + m) * KVLEN + t) * HIDD + hcol] = (__bf16)hv;
        if (t == S_LEN - 1) {
          hnf[(size_t)(b0 + m) * HIDD + hcol] = hv;
          hnb[(size_t)(b0 + m) * HIDD + hcol] = (__bf16)hv;
        }
      }
    }

    // prime x_{t+1}
    if (t + 1 < S_LEN) {
      const int m = tid >> 5, kk = (tid & 31) << 1;
      const float* sp = src + ((size_t)(b0 + m) * S_LEN + (t + 1)) * 64 + kk;
      As[m * KC + kk]     = (__bf16)sp[0];
      As[m * KC + kk + 1] = (__bf16)sp[1];
    }
    __syncthreads();   // A ready for step t+1
  }
}

// ---------------------------------------------------------------------------
// Generic WMMA GEMM: C[M,N] = concat_K(A1[M,K1], A2[M,K2]) @ Bt^T (+bias)
// Bt is N-major bf16 [N][K1+K2]. Block = 256 threads (8 waves, one 16x16
// output tile each); grid = (M/16, N/128).
// ---------------------------------------------------------------------------
__global__ __launch_bounds__(256)
void gemm_kernel(const __bf16* __restrict__ A1, int K1,
                 const __bf16* __restrict__ A2, int K2,
                 const __bf16* __restrict__ Bt, const float* __restrict__ bias,
                 float* __restrict__ Cf, __bf16* __restrict__ Cbf, int N) {
  const int tid  = threadIdx.x;
  const int w    = tid >> 5;
  const int ln   = tid & 31;
  const int ln16 = ln & 15;
  const int hh   = ln >> 4;
  const int mb   = blockIdx.x * 16;
  const int nb   = (blockIdx.y * 8 + w) * 16;
  const int Ktot = K1 + K2;
  const int m    = mb + ln16;
  const __bf16* brow = Bt + (size_t)(nb + ln16) * Ktot;

  v8f acc = {};
  for (int k0 = 0; k0 < Ktot; k0 += 32) {
    const int c1 = k0 + hh * 8;
    const int c2 = c1 + 16;
    // K1 is a multiple of 32, so 8-wide chunks never straddle the seam
    const __bf16* ap1 = (c1 < K1) ? (A1 + (size_t)m * K1 + c1)
                                  : (A2 + (size_t)m * K2 + (c1 - K1));
    const __bf16* ap2 = (c2 < K1) ? (A1 + (size_t)m * K1 + c2)
                                  : (A2 + (size_t)m * K2 + (c2 - K1));
    v16bf a = ldfrag(ap1, ap2);
    v16bf b = ldfrag(brow + c1, brow + c2);
    acc = wmma_bf16(a, b, acc);
  }
  const float bv = bias ? bias[nb + ln16] : 0.0f;
#pragma unroll
  for (int r = 0; r < 8; ++r) {
    const int gm = mb + r + (hh << 3);
    const int gn = nb + ln16;
    const float v = acc[r] + bv;
    if (Cf)  Cf [(size_t)gm * N + gn] = v;
    if (Cbf) Cbf[(size_t)gm * N + gn] = (__bf16)v;
  }
}

// ---------------------------------------------------------------------------
// Causal conv (K=3, dil=1) on enc as a shifted GEMM:
// ck[b,t,o] = sum_tau sum_i enc[b, t-2+tau, i] * conv_w[o,i,tau] + conv_b[o]
// M = 64*320 rows (tiles never straddle a batch: 320 % 16 == 0).
// ---------------------------------------------------------------------------
__global__ __launch_bounds__(256)
void conv_kernel(const __bf16* __restrict__ enc, const __bf16* __restrict__ cwB,
                 const float* __restrict__ conv_b, __bf16* __restrict__ ck) {
  const int tid  = threadIdx.x;
  const int w    = tid >> 5;
  const int ln   = tid & 31;
  const int ln16 = ln & 15;
  const int hh   = ln >> 4;
  const int mb   = blockIdx.x * 16;
  const int nb   = (blockIdx.y * 8 + w) * 16;
  const int m    = mb + ln16;
  const int b    = m / KVLEN;
  const int t    = m % KVLEN;
  const __bf16* brow = cwB + (size_t)(nb + ln16) * (3 * HIDD);

  v8f acc = {};
  for (int tau = 0; tau < 3; ++tau) {
    const int tt = t - 2 + tau;                       // causal shift (per lane)
    const __bf16* arow = enc + ((size_t)b * KVLEN + tt) * HIDD;
    for (int k0 = 0; k0 < HIDD; k0 += 32) {
      const int c1 = k0 + hh * 8;
      const int c2 = c1 + 16;
      v16bf a = {};
      if (tt >= 0) a = ldfrag(arow + c1, arow + c2);  // padded rows -> 0
      const int kb = tau * HIDD + k0;
      v16bf bf = ldfrag(brow + kb + hh * 8, brow + kb + 16 + hh * 8);
      acc = wmma_bf16(a, bf, acc);
    }
  }
  const float bv = conv_b[nb + ln16];
#pragma unroll
  for (int r = 0; r < 8; ++r) {
    const int gm = mb + r + (hh << 3);
    ck[(size_t)gm * HIDD + nb + ln16] = (__bf16)(acc[r] + bv);
  }
}

// ---------------------------------------------------------------------------
// q-side causal conv: only tap tau=2 survives (Lq == 1, pad == 2)
// cq[b,o] = sum_i h_n[b,i]*conv_w[o,i,2] + conv_b[o]
// ---------------------------------------------------------------------------
__global__ __launch_bounds__(512)
void cq_kernel(const float* __restrict__ hnf, const float* __restrict__ cw,
               const float* __restrict__ cb, __bf16* __restrict__ cq) {
  const int b = blockIdx.x, o = threadIdx.x;
  float s = cb[o];
  const float* hr = hnf + (size_t)b * HIDD;
  for (int i = 0; i < HIDD; ++i)
    s += hr[i] * cw[(o * HIDD + i) * 3 + 2];
  cq[(size_t)b * HIDD + o] = (__bf16)s;
}

// ---------------------------------------------------------------------------
// Single-query attention per (b, head): scores -> softmax -> ctx  (~20 MFLOP)
// ---------------------------------------------------------------------------
__global__ __launch_bounds__(KVLEN)
void attn_kernel(const float* __restrict__ Q, const float* __restrict__ K,
                 const float* __restrict__ V, float* __restrict__ ctx) {
  __shared__ float sc[KVLEN];
  __shared__ float red[2];
  const int bh = blockIdx.x;
  const int b = bh >> 3, h = bh & 7;
  const int t = threadIdx.x;

  const float* q  = Q + (size_t)b * HIDD + h * DH;
  const float* kr = K + ((size_t)b * KVLEN + t) * HIDD + h * DH;
  float s = 0.0f;
  for (int d = 0; d < DH; ++d) s += q[d] * kr[d];
  sc[t] = s * 0.125f;                       // 1/sqrt(64)
  __syncthreads();
  if (t == 0) {
    float mx = sc[0];
    for (int i = 1; i < KVLEN; ++i) mx = fmaxf(mx, sc[i]);
    red[0] = mx;
  }
  __syncthreads();
  sc[t] = __expf(sc[t] - red[0]);
  __syncthreads();
  if (t == 0) {
    float sm = 0.0f;
    for (int i = 0; i < KVLEN; ++i) sm += sc[i];
    red[1] = sm;
  }
  __syncthreads();
  if (t < DH) {
    float a = 0.0f;
    for (int tt = 0; tt < KVLEN; ++tt)
      a += sc[tt] * V[((size_t)b * KVLEN + tt) * HIDD + h * DH + t];
    ctx[(size_t)b * HIDD + h * DH + t] = a / red[1];
  }
}

// ---------------------------------------------------------------------------
// attn_out = ctx @ Wfc; h_after = [h_n | attn_out] @ cat_W + cat_b;
// out[b] = h_after @ out_W
// ---------------------------------------------------------------------------
__global__ __launch_bounds__(256)
void final_kernel(const float* __restrict__ ctx, const float* __restrict__ hnf,
                  const float* __restrict__ Wfc, const float* __restrict__ catW,
                  const float* __restrict__ catb, const float* __restrict__ outW,
                  float* __restrict__ out) {
  __shared__ float ao[HIDD];
  __shared__ float ha[64];
  const int b = blockIdx.x, tid = threadIdx.x;
  for (int o = tid; o < HIDD; o += 256) {
    float s = 0.0f;
    for (int j = 0; j < HIDD; ++j)
      s += ctx[(size_t)b * HIDD + j] * Wfc[(size_t)j * HIDD + o];
    ao[o] = s;
  }
  __syncthreads();
  if (tid < 64) {
    float s = catb[tid];
    for (int j = 0; j < HIDD; ++j) s += hnf[(size_t)b * HIDD + j] * catW[j * 64 + tid];
    for (int j = 0; j < HIDD; ++j) s += ao[j] * catW[(HIDD + j) * 64 + tid];
    ha[tid] = s;
  }
  __syncthreads();
  if (tid == 0) {
    float s = 0.0f;
    for (int p = 0; p < 64; ++p) s += ha[p] * outW[p];
    out[b] = s;
  }
}

// ---------------------------------------------------------------------------
// Host launch
// ---------------------------------------------------------------------------
extern "C" void kernel_launch(void* const* d_in, const int* in_sizes, int n_in,
                              void* d_out, int out_size, void* d_ws, size_t ws_size,
                              hipStream_t stream) {
  const float* src    = (const float*)d_in[0];
  // d_in[1] = tgt (only its length matters: kv_len = 256 + 64 = 320)
  const float* W_ih   = (const float*)d_in[2];
  const float* W_hh   = (const float*)d_in[3];
  const float* b_ih   = (const float*)d_in[4];
  const float* b_hh   = (const float*)d_in[5];
  const float* conv_w = (const float*)d_in[6];
  const float* conv_b = (const float*)d_in[7];
  const float* Wq     = (const float*)d_in[8];
  const float* Wk     = (const float*)d_in[9];
  const float* Wv     = (const float*)d_in[10];
  const float* Wfc    = (const float*)d_in[11];
  const float* cat_W  = (const float*)d_in[12];
  const float* cat_b  = (const float*)d_in[13];
  const float* out_W  = (const float*)d_in[14];
  float* outp = (float*)d_out;

  char* ws = (char*)d_ws;
  size_t off = 0;
  auto alloc = [&](size_t bytes) -> char* {
    char* p = ws + off;
    off = (off + bytes + 255) & ~(size_t)255;
    return p;
  };

  __bf16* Wc    = (__bf16*)alloc((size_t)GATES * KC * 2);
  float*  biasc = (float*) alloc((size_t)GATES * 4);
  __bf16* cwB   = (__bf16*)alloc((size_t)HIDD * 3 * HIDD * 2);
  __bf16* WkT   = (__bf16*)alloc((size_t)HIDD * 1024 * 2);
  __bf16* WqT   = (__bf16*)alloc((size_t)HIDD * 1024 * 2);
  __bf16* WvT   = (__bf16*)alloc((size_t)HIDD * HIDD * 2);
  __bf16* enc   = (__bf16*)alloc((size_t)B_TOT * KVLEN * HIDD * 2);
  __bf16* ck    = (__bf16*)alloc((size_t)B_TOT * KVLEN * HIDD * 2);
  float*  Kf    = (float*) alloc((size_t)B_TOT * KVLEN * HIDD * 4);
  float*  Vf    = (float*) alloc((size_t)B_TOT * KVLEN * HIDD * 4);
  float*  hnf   = (float*) alloc((size_t)B_TOT * HIDD * 4);
  __bf16* hnb   = (__bf16*)alloc((size_t)B_TOT * HIDD * 2);
  __bf16* cq    = (__bf16*)alloc((size_t)B_TOT * HIDD * 2);
  float*  Qf    = (float*) alloc((size_t)B_TOT * HIDD * 4);
  float*  ctx   = (float*) alloc((size_t)B_TOT * HIDD * 4);
  (void)ws_size; (void)in_sizes; (void)n_in; (void)out_size;

  // weight repacks (bf16, N-major)
  prep_wc  <<<(GATES * KC + 255) / 256, 256, 0, stream>>>(W_ih, W_hh, b_ih, b_hh, Wc, biasc);
  prep_conv<<<(HIDD * 3 * HIDD + 255) / 256, 256, 0, stream>>>(conv_w, cwB);
  prep_T   <<<(HIDD * 1024 + 255) / 256, 256, 0, stream>>>(Wk, WkT, 1024);
  prep_T   <<<(HIDD * 1024 + 255) / 256, 256, 0, stream>>>(Wq, WqT, 1024);
  prep_vT  <<<(HIDD * HIDD + 255) / 256, 256, 0, stream>>>(Wv, WvT);

  // persistent LSTM recurrence (4 batch-tiles x 16 waves)
  lstm_kernel<<<4, 512, 0, stream>>>(src, Wc, biasc, enc, hnf, hnb);

  const int Mrows = B_TOT * KVLEN;            // 20480
  // ck = causal_conv(enc)
  conv_kernel<<<dim3(Mrows / 16, 4), 256, 0, stream>>>(enc, cwB, conv_b, ck);
  // V = enc @ (Wv_top + Wv_bot)
  gemm_kernel<<<dim3(Mrows / 16, 4), 256, 0, stream>>>(enc, HIDD, (const __bf16*)nullptr, 0,
                                                       WvT, nullptr, Vf, nullptr, HIDD);
  // K = [enc | ck] @ Wk
  gemm_kernel<<<dim3(Mrows / 16, 4), 256, 0, stream>>>(enc, HIDD, ck, HIDD,
                                                       WkT, nullptr, Kf, nullptr, HIDD);
  // q-side conv, then Q = [h_n | cq] @ Wq
  cq_kernel<<<B_TOT, HIDD, 0, stream>>>(hnf, conv_w, conv_b, cq);
  gemm_kernel<<<dim3(B_TOT / 16, 4), 256, 0, stream>>>(hnb, HIDD, cq, HIDD,
                                                       WqT, nullptr, Qf, nullptr, HIDD);
  // attention (Lq = 1) and output head
  attn_kernel<<<B_TOT * NHEAD, KVLEN, 0, stream>>>(Qf, Kf, Vf, ctx);
  final_kernel<<<B_TOT, 256, 0, stream>>>(ctx, hnf, Wfc, cat_W, cat_b, out_W, outp);
}
